// CharEmbedding_66881230733377
// MI455X (gfx1250) — matrix operand order repaired and checked by reading, not compiled, not run on hardware
//
#include <hip/hip_runtime.h>
#include <hip/hip_bf16.h>
#include <math.h>

typedef __attribute__((ext_vector_type(8)))  float  v8f;
typedef __attribute__((ext_vector_type(16))) __bf16 v16bf;
typedef __attribute__((ext_vector_type(8)))  __bf16 v8bf;

#define B_   64
#define LW_  512
#define LC_  4096
#define DW_  300
#define DC_  64
#define H_   128
#define M_   (B_ * LW_)   /* 32768 */
#define KP_  384          /* 364 padded to multiple of 32 */
#define SXS  392          /* sX LDS row stride (elements): 784B = 196 dw, 196%64=4 -> no bank conflicts */
#define SHS  136          /* sH LDS row stride (elements): 272B = 68 dw,  68%64=4 -> no bank conflicts */

// ---------------------------------------------------------------------------
// Weight prep: pad+convert W_proj [128,364] -> bf16 [128,384]; HxH -> bf16
// ---------------------------------------------------------------------------
__global__ void k_pad_wproj(const float* __restrict__ w, __bf16* __restrict__ out) {
    int idx = blockIdx.x * blockDim.x + threadIdx.x;   // < 128*384
    int n = idx / KP_;
    int k = idx - n * KP_;
    float v = (k < (DW_ + DC_)) ? w[n * (DW_ + DC_) + k] : 0.0f;
    out[idx] = (__bf16)v;
}

__global__ void k_cvt_bf16(const float* __restrict__ src, __bf16* __restrict__ dst, int n) {
    int i = blockIdx.x * blockDim.x + threadIdx.x;
    if (i < n) dst[i] = (__bf16)src[i];
}

// ---------------------------------------------------------------------------
// WMMA fragment loaders per CDNA5 ISA 16-bit VGPR layouts (wave32).
// A (16x32 MxK): lane l -> row m=l&15; hi=l>>4; K halves = hi*8+0..7 , 16+hi*8+0..7
// B (32x16 KxN): lane l -> col n=l&15; hi=l>>4; K halves = hi*8+0..7 , 16+hi*8+0..7
// Works for both LDS-resident (ds_load_b128) and global rows.
// ---------------------------------------------------------------------------
__device__ __forceinline__ v16bf load_fragA(const __bf16* row, int kb, int hi) {
    v8bf a0 = *(const v8bf*)(row + kb + hi * 8);
    v8bf a1 = *(const v8bf*)(row + kb + 16 + hi * 8);
    return __builtin_shufflevector(a0, a1, 0,1,2,3,4,5,6,7,8,9,10,11,12,13,14,15);
}
__device__ __forceinline__ v16bf load_fragB(const __bf16* row, int kb, int hi) {
    v8bf b0 = *(const v8bf*)(row + kb + hi * 8);
    v8bf b1 = *(const v8bf*)(row + kb + 16 + hi * 8);
    return __builtin_shufflevector(b0, b1, 0,1,2,3,4,5,6,7,8,9,10,11,12,13,14,15);
}
#define WMMA_BF16(a, b, c) __builtin_amdgcn_wmma_f32_16x16x32_bf16(false, (a), false, (b), (short)0, (c), false, false)

// ---------------------------------------------------------------------------
// Fully fused kernel: one block = 16 rows of x, 8 waves = 8 N-tiles of 16.
//   LDS stage sX[16,384]bf16 (word gather + char pooling + pad)
//   -> proj WMMA (A: LDS, B: global bf16 weights), x stays fp32 in registers
//   -> 2x highway layers (bf16 x via LDS sH, residual from registers)
//   -> fp32 store to d_out.
// No fp32/bf16 activation ever round-trips through global memory.
// ---------------------------------------------------------------------------
__global__ void __launch_bounds__(256) k_fused(
        const int*    __restrict__ w,   const float* __restrict__ wv,
        const int*    __restrict__ c,   const int*   __restrict__ p,
        const float*  __restrict__ ce,
        const __bf16* __restrict__ Wpb,
        const __bf16* __restrict__ Wg0b, const float* __restrict__ bg0,
        const __bf16* __restrict__ Wt0b, const float* __restrict__ bt0,
        const __bf16* __restrict__ Wg1b, const float* __restrict__ bg1,
        const __bf16* __restrict__ Wt1b, const float* __restrict__ bt1,
        float* __restrict__ out)
{
    __shared__ __bf16 sX[16 * SXS];
    __shared__ __bf16 sH[16 * SHS];
    __shared__ int sLo[16], sHi[16];

    const int t     = threadIdx.x;
    const int mbase = blockIdx.x * 16;
    const int b     = mbase >> 9;            // batch row; blocks never straddle (512%16==0)

    // ---- phase 1: word-vector gather (fp32->bf16) + K-pad + binary search ----
    {
        int r = t >> 4, l16 = t & 15;
        const float* wvrow = wv + (size_t)w[mbase + r] * DW_;
        for (int chunk = l16; chunk < DW_ / 4; chunk += 16) {       // 75 float4 chunks
            float4 v4 = *(const float4*)(wvrow + chunk * 4);
            __bf16* dst = &sX[r * SXS + chunk * 4];
            dst[0] = (__bf16)v4.x; dst[1] = (__bf16)v4.y;
            dst[2] = (__bf16)v4.z; dst[3] = (__bf16)v4.w;
        }
        for (int k = DW_ + DC_ + l16; k < KP_; k += 16)             // zero pad 364..383
            sX[r * SXS + k] = (__bf16)0.0f;
    }
    if (t < 16) {   // p is sorted per batch row: [lo,hi) char range for word i
        const int* prow = p + (size_t)b * LC_;
        int i = (mbase + t) & (LW_ - 1);
        int lo = 0, hi = LC_;
        while (lo < hi) { int mid = (lo + hi) >> 1; if (prow[mid] < i) lo = mid + 1; else hi = mid; }
        int lo2 = lo, hi2 = LC_;
        while (lo2 < hi2) { int mid = (lo2 + hi2) >> 1; if (prow[mid] < i + 1) lo2 = mid + 1; else hi2 = mid; }
        sLo[t] = lo; sHi[t] = lo2;
    }
    __syncthreads();

    // ---- phase 2: char pooling, 16 threads/row x 4 dims each -> sX[:,300:364] ----
    {
        int r = t >> 4;
        int d = (t & 15) * 4;
        int lo = sLo[r], hi = sHi[r];
        const int* crow = c + (size_t)b * LC_;
        float s0 = 0.f, s1 = 0.f, s2 = 0.f, s3 = 0.f;
        for (int j = lo; j < hi; ++j) {
            int cc = crow[j];
            cc = (cc == 32) ? 0 : cc;          // space -> padding row 0
            cc = (cc > 256) ? 256 : cc;        // clamp to UNK row
            float4 e = *(const float4*)(ce + cc * DC_ + d);
            s0 += e.x; s1 += e.y; s2 += e.z; s3 += e.w;
        }
        float inv = 1.0f / (0.001f + sqrtf((float)(hi - lo)));
        __bf16* dst = &sX[r * SXS + DW_ + d];
        dst[0] = (__bf16)(s0 * inv); dst[1] = (__bf16)(s1 * inv);
        dst[2] = (__bf16)(s2 * inv); dst[3] = (__bf16)(s3 * inv);
    }
    __syncthreads();

    const int lane  = t & 31;
    const int wave  = t >> 5;
    const int hi16  = lane >> 4;
    const int l15   = lane & 15;
    const int nbase = wave * 16;

    // ---- projection GEMM: A fragments from LDS, B from global bf16 weights ----
    v8f xcur = {};
    {
        const __bf16* arow = &sX[l15 * SXS];
        const __bf16* brow = Wpb + (size_t)(nbase + l15) * KP_;
#pragma unroll
        for (int kc = 0; kc < KP_ / 32; ++kc) {
            int kb = kc * 32;
            v16bf av = load_fragA(arow, kb, hi16);
            v16bf bv = load_fragB(brow, kb, hi16);
            xcur = WMMA_BF16(av, bv, xcur);
        }
    }

    // ---- 2 fused highway layers; residual x lives in registers ----
    const __bf16* Wg[2] = { Wg0b, Wg1b };
    const float*  bg[2] = { bg0,  bg1  };
    const __bf16* Wt[2] = { Wt0b, Wt1b };
    const float*  bt[2] = { bt0,  bt1  };

    for (int layer = 0; layer < 2; ++layer) {
        // spill bf16(x) tile to LDS: wave w owns cols 16w..16w+15, rows via r+hi16*8
#pragma unroll
        for (int r = 0; r < 8; ++r) {
            int row = r + hi16 * 8;
            sH[row * SHS + nbase + l15] = (__bf16)xcur[r];
        }
        __syncthreads();

        v8f ag = {}, at = {};
        const __bf16* arow = &sH[l15 * SHS];
        const __bf16* grow = Wg[layer] + (size_t)(nbase + l15) * H_;
        const __bf16* trow = Wt[layer] + (size_t)(nbase + l15) * H_;
#pragma unroll
        for (int kc = 0; kc < H_ / 32; ++kc) {
            int kb = kc * 32;
            v16bf av = load_fragA(arow, kb, hi16);   // shared A fragment
            v16bf gv = load_fragB(grow, kb, hi16);
            v16bf tv = load_fragB(trow, kb, hi16);
            ag = WMMA_BF16(av, gv, ag);
            at = WMMA_BF16(av, tv, at);
        }
        __syncthreads();   // all sH reads done before next layer overwrites

        float bgv = bg[layer][nbase + l15];   // output col is constant per lane
        float btv = bt[layer][nbase + l15];
#pragma unroll
        for (int r = 0; r < 8; ++r) {
            float g  = 1.0f / (1.0f + __expf(-(ag[r] + bgv)));
            float tt = fmaxf(at[r] + btv, 0.0f);
            xcur[r]  = g * tt + (1.0f - g) * xcur[r];
        }
    }

    // ---- final fp32 store ----
#pragma unroll
    for (int r = 0; r < 8; ++r) {
        int row = mbase + r + hi16 * 8;
        out[(size_t)row * H_ + nbase + l15] = xcur[r];
    }
}

// ---------------------------------------------------------------------------
// Host launcher
// ---------------------------------------------------------------------------
extern "C" void kernel_launch(void* const* d_in, const int* in_sizes, int n_in,
                              void* d_out, int out_size, void* d_ws, size_t ws_size,
                              hipStream_t stream) {
    (void)in_sizes; (void)n_in; (void)out_size; (void)ws_size;

    const int*   w   = (const int*)  d_in[0];   // [B,LW]
    const int*   c   = (const int*)  d_in[1];   // [B,LC]
    const int*   p   = (const int*)  d_in[2];   // [B,LC] sorted per row
    const float* wv  = (const float*)d_in[3];   // [V,DW]
    const float* ce  = (const float*)d_in[4];   // [257,DC]
    const float* Wp  = (const float*)d_in[5];   // [H, DW+DC]
    const float* Wt0 = (const float*)d_in[6];
    const float* bt0 = (const float*)d_in[7];
    const float* Wg0 = (const float*)d_in[8];
    const float* bg0 = (const float*)d_in[9];
    const float* Wt1 = (const float*)d_in[10];
    const float* bt1 = (const float*)d_in[11];
    const float* Wg1 = (const float*)d_in[12];
    const float* bg1 = (const float*)d_in[13];
    float* out = (float*)d_out;                 // [B,LW,H] fp32

    // Workspace: only bf16 weight copies (~230 KB)
    char* ws = (char*)d_ws;
    const size_t WPB_B = (size_t)H_ * KP_ * 2;  // 98,304
    const size_t WHB_B = (size_t)H_ * H_  * 2;  // 32,768
    __bf16* Wpb  = (__bf16*)(ws);
    __bf16* Wt0b = (__bf16*)(ws + WPB_B);
    __bf16* Wg0b = (__bf16*)(ws + WPB_B + 1 * WHB_B);
    __bf16* Wt1b = (__bf16*)(ws + WPB_B + 2 * WHB_B);
    __bf16* Wg1b = (__bf16*)(ws + WPB_B + 3 * WHB_B);

    k_pad_wproj<<<(H_ * KP_) / 256, 256, 0, stream>>>(Wp, Wpb);
    k_cvt_bf16 <<<(H_ * H_ + 255) / 256, 256, 0, stream>>>(Wt0, Wt0b, H_ * H_);
    k_cvt_bf16 <<<(H_ * H_ + 255) / 256, 256, 0, stream>>>(Wg0, Wg0b, H_ * H_);
    k_cvt_bf16 <<<(H_ * H_ + 255) / 256, 256, 0, stream>>>(Wt1, Wt1b, H_ * H_);
    k_cvt_bf16 <<<(H_ * H_ + 255) / 256, 256, 0, stream>>>(Wg1, Wg1b, H_ * H_);

    k_fused<<<M_ / 16, 256, 0, stream>>>(w, wv, c, p, ce, Wpb,
                                         Wg0b, bg0, Wt0b, bt0,
                                         Wg1b, bg1, Wt1b, bt1, out);
}
